// AFNO2D_41394894799854
// MI455X (gfx1250) — matrix operand order repaired and checked by reading, not compiled, not run on hardware
//
#include <hip/hip_runtime.h>
#include <math.h>

#define BB 4
#define CC 128
#define HH 256
#define WW 256
#define WF 129
#define NPOS (BB*HH*WF)      // 132096 frequency positions
#define NTILE (NPOS/16)      // 8256 tiles of 16 positions
#define TS 17                // padded LDS row stride (bank-conflict avoidance)

typedef __attribute__((ext_vector_type(16))) _Float16 v16h;
typedef __attribute__((ext_vector_type(8)))  float    v8f;

// ---------------------------------------------------------------------------
// 256-point radix-2 DIT FFT in LDS. 128 threads per transform (t = 0..127).
// sign = -1 forward, +1 inverse (unnormalized).
// ---------------------------------------------------------------------------
__device__ __forceinline__ void fft256(float* re, float* im, int t, float sign) {
    __syncthreads();                       // cover preceding LDS fills
    #pragma unroll
    for (int q = 0; q < 2; ++q) {          // bit-reversal permutation
        int j  = t + (q << 7);
        int rj = __brev(j) >> 24;
        if (rj > j) {
            float ar = re[j],  ai = im[j];
            float br = re[rj], bi = im[rj];
            re[j]  = br; im[j]  = bi;
            re[rj] = ar; im[rj] = ai;
        }
    }
    __syncthreads();
    #pragma unroll
    for (int len = 2; len <= 256; len <<= 1) {
        const int half = len >> 1;
        int grp = t / half, p = t % half;
        int i1 = grp * len + p, i2 = i1 + half;
        float ang = sign * 6.28318530717958647692f * (float)p / (float)len;
        float sn, cs;
        __sincosf(ang, &sn, &cs);
        float br = re[i2], bi = im[i2];
        float tr = br * cs - bi * sn;
        float ti = br * sn + bi * cs;
        float ar = re[i1], ai = im[i1];
        re[i1] = ar + tr; im[i1] = ai + ti;
        re[i2] = ar - tr; im[i2] = ai - ti;
        __syncthreads();
    }
}

// ---------------------------------------------------------------------------
// Forward row rFFT along W: x[b][c][h][0..255] -> spectrum[pos][c] (wf=0..128)
// Layout: pos = (b*256 + h)*129 + wf, channel innermost (stride 128 floats).
// ---------------------------------------------------------------------------
__global__ __launch_bounds__(256) void fwd_row_fft(const float* __restrict__ x,
                                                   float* __restrict__ gRe,
                                                   float* __restrict__ gIm) {
    __shared__ float re[2][256], im[2][256];
    int sub = threadIdx.x >> 7, t = threadIdx.x & 127;
    int row = blockIdx.x * 2 + sub;                 // (b*128 + c)*256 + h
    const float* xr = x + (size_t)row * 256;
    re[sub][t]       = xr[t];       im[sub][t]       = 0.f;
    re[sub][t + 128] = xr[t + 128]; im[sub][t + 128] = 0.f;
    fft256(re[sub], im[sub], t, -1.f);
    int h = row & 255, c = (row >> 8) & 127, b = row >> 15;
    size_t pb = (size_t)(b * 256 + h) * 129;
    size_t o  = (pb + t) * 128 + c;
    gRe[o] = re[sub][t]; gIm[o] = im[sub][t];
    if (t == 0) {
        size_t o2 = (pb + 128) * 128 + c;
        gRe[o2] = re[sub][128]; gIm[o2] = im[sub][128];
    }
}

// ---------------------------------------------------------------------------
// Column FFT along H (in-place, full complex 256-pt). sign=-1 fwd / +1 inv.
// ---------------------------------------------------------------------------
__global__ __launch_bounds__(256) void col_fft(float* __restrict__ gRe,
                                               float* __restrict__ gIm,
                                               float sign, float scale) {
    __shared__ float re[2][256], im[2][256];
    int sub = threadIdx.x >> 7, t = threadIdx.x & 127;
    int q = blockIdx.x * 2 + sub;                   // (b*129 + wf)*128 + c
    int c = q & 127;
    int rem = q >> 7;
    int wf = rem % 129, b = rem / 129;
    #pragma unroll
    for (int qq = 0; qq < 2; ++qq) {
        int h = t + (qq << 7);
        size_t a = ((size_t)(b * 256 + h) * 129 + wf) * 128 + c;
        re[sub][h] = gRe[a]; im[sub][h] = gIm[a];
    }
    fft256(re[sub], im[sub], t, sign);
    #pragma unroll
    for (int qq = 0; qq < 2; ++qq) {
        int h = t + (qq << 7);
        size_t a = ((size_t)(b * 256 + h) * 129 + wf) * 128 + c;
        gRe[a] = re[sub][h] * scale;
        gIm[a] = im[sub][h] * scale;
    }
}

// ---------------------------------------------------------------------------
// Block-diagonal complex MLP on the spectrum via WMMA (in-place).
// Each wave: one 16-position tile of one channel block k.
// Complex matmuls folded into K=32:  [Xr|Xi] x [Wr;-Wi] and [Xr|Xi] x [Wi;Wr].
// ---------------------------------------------------------------------------
__global__ __launch_bounds__(256) void afno_wmma_kernel(
        float* __restrict__ gRe, float* __restrict__ gIm,
        const float* __restrict__ w1, const float* __restrict__ b1,
        const float* __restrict__ w2, const float* __restrict__ b2) {
    __shared__ float smem[8 * 4 * 16 * TS];
    const int wave = threadIdx.x >> 5;
    const int lane = threadIdx.x & 31;
    const int n    = lane & 15;          // N column (and A's M row)
    const int hi   = lane >> 4;          // lane half
    const int mo   = hi << 3;            // row offset for C/D layout
    const int k    = blockIdx.x / (NTILE / 8);
    const int tile = (blockIdx.x % (NTILE / 8)) * 8 + wave;
    const size_t tb = (size_t)tile * 16;
    const int c0 = k * 16;

    float* Lre  = smem + wave * (4 * 16 * TS);
    float* Lim  = Lre  + 16 * TS;
    float* LO1r = Lim  + 16 * TS;
    float* LO1i = LO1r + 16 * TS;

    // stage 16x16 complex tile into LDS (float4 coalesced)
    {
        const float* src = (hi ? gIm : gRe) + (tb + n) * 128 + c0;
        float* dst = (hi ? Lim : Lre) + n * TS;
        #pragma unroll
        for (int j = 0; j < 16; j += 4) {
            float4 v = *(const float4*)(src + j);
            dst[j] = v.x; dst[j + 1] = v.y; dst[j + 2] = v.z; dst[j + 3] = v.w;
        }
    }

    // pack B matrices in contiguous k order (k = 16*half + i): VGPR j holds rows 2j,2j+1,
    // lane half selects K or K+16  ->  [W_r ; -W_i] and [W_i ; W_r]
    const float* W1r = w1 + k * 256;
    const float* W1i = w1 + 2048 + k * 256;
    const float* W2r = w2 + k * 256;
    const float* W2i = w2 + 2048 + k * 256;
    v16h B1r, B1i, B2r, B2i;
    #pragma unroll
    for (int i = 0; i < 16; ++i) {
        float a = W1r[i * 16 + n], bw = W1i[i * 16 + n];
        float c = W2r[i * 16 + n], d  = W2i[i * 16 + n];
        B1r[i] = (_Float16)(hi ? -bw : a);
        B1i[i] = (_Float16)(hi ?  a  : bw);
        B2r[i] = (_Float16)(hi ? -d  : c);
        B2i[i] = (_Float16)(hi ?  c  : d);
    }
    const float bias1r = b1[k * 16 + n], bias1i = b1[128 + k * 16 + n];
    const float bias2r = b2[k * 16 + n], bias2i = b2[128 + k * 16 + n];

    __syncthreads();

    // A = [Xre | Xim], 16x32 f16: lanes<16 -> K 0-15 (Xre), lanes>=16 -> K 16-31 (Xim)
    v16h A;
    #pragma unroll
    for (int j = 0; j < 8; ++j) {
        A[j]     = (_Float16)(hi ? Lim : Lre)[n * TS + j];
        A[8 + j] = (_Float16)(hi ? Lim : Lre)[n * TS + 8 + j];
    }

    v8f zero = {};
    v8f c1r = __builtin_amdgcn_wmma_f32_16x16x32_f16(false, A, false, B1r, (short)0, zero, false, false);
    v8f c1i = __builtin_amdgcn_wmma_f32_16x16x32_f16(false, A, false, B1i, (short)0, zero, false, false);

    #pragma unroll
    for (int r = 0; r < 8; ++r) {
        LO1r[(r + mo) * TS + n] = fmaxf(c1r[r] + bias1r, 0.f);
        LO1i[(r + mo) * TS + n] = fmaxf(c1i[r] + bias1i, 0.f);
    }
    __syncthreads();

    v16h A2;
    #pragma unroll
    for (int j = 0; j < 8; ++j) {
        A2[j]     = (_Float16)(hi ? LO1i : LO1r)[n * TS + j];
        A2[8 + j] = (_Float16)(hi ? LO1i : LO1r)[n * TS + 8 + j];
    }
    v8f c2r = __builtin_amdgcn_wmma_f32_16x16x32_f16(false, A2, false, B2r, (short)0, zero, false, false);
    v8f c2i = __builtin_amdgcn_wmma_f32_16x16x32_f16(false, A2, false, B2i, (short)0, zero, false, false);

    // bias + softshrink + complex multiply with original spectrum, write back
    #pragma unroll
    for (int r = 0; r < 8; ++r) {
        float o2r = c2r[r] + bias2r;
        float o2i = c2i[r] + bias2i;
        float sr = copysignf(fmaxf(fabsf(o2r) - 0.01f, 0.f), o2r);
        float si = copysignf(fmaxf(fabsf(o2i) - 0.01f, 0.f), o2i);
        float xr = Lre[(r + mo) * TS + n];
        float xi = Lim[(r + mo) * TS + n];
        size_t o = (tb + r + mo) * 128 + c0 + n;
        gRe[o] = sr * xr - si * xi;
        gIm[o] = sr * xi + si * xr;
    }
}

// ---------------------------------------------------------------------------
// Inverse row FFT along W with Hermitian reconstruction + residual add.
// ---------------------------------------------------------------------------
__global__ __launch_bounds__(256) void inv_row_fft(const float* __restrict__ gRe,
                                                   const float* __restrict__ gIm,
                                                   const float* __restrict__ x,
                                                   float* __restrict__ out) {
    __shared__ float re[2][256], im[2][256];
    int sub = threadIdx.x >> 7, t = threadIdx.x & 127;
    int row = blockIdx.x * 2 + sub;                 // (b*128 + c)*256 + h
    int h = row & 255, c = (row >> 8) & 127, b = row >> 15;
    size_t pb = (size_t)(b * 256 + h) * 129;
    {
        size_t a = (pb + t) * 128 + c;              // wf = t (0..127)
        re[sub][t] = gRe[a]; im[sub][t] = gIm[a];
        int wf = (t == 0) ? 128 : (128 - t);        // j2 = t+128 -> mirror
        size_t a2 = (pb + wf) * 128 + c;
        float s = (t == 0) ? 1.f : -1.f;            // conjugate mirror
        re[sub][t + 128] = gRe[a2];
        im[sub][t + 128] = s * gIm[a2];
    }
    fft256(re[sub], im[sub], t, +1.f);
    const float* xr = x + (size_t)row * 256;
    float* orow = out + (size_t)row * 256;
    orow[t]       = xr[t]       + re[sub][t]       * (1.f / 256.f);
    orow[t + 128] = xr[t + 128] + re[sub][t + 128] * (1.f / 256.f);
}

// ---------------------------------------------------------------------------
extern "C" void kernel_launch(void* const* d_in, const int* in_sizes, int n_in,
                              void* d_out, int out_size, void* d_ws, size_t ws_size,
                              hipStream_t stream) {
    (void)in_sizes; (void)n_in; (void)out_size; (void)ws_size;
    const float* x  = (const float*)d_in[0];
    const float* w1 = (const float*)d_in[1];
    const float* b1 = (const float*)d_in[2];
    const float* w2 = (const float*)d_in[3];
    const float* b2 = (const float*)d_in[4];
    float* out = (float*)d_out;
    float* gRe = (float*)d_ws;                       // [NPOS][128] re plane
    float* gIm = gRe + (size_t)NPOS * 128;           // [NPOS][128] im plane

    dim3 blk(256);
    fwd_row_fft<<<BB * CC * HH / 2, blk, 0, stream>>>(x, gRe, gIm);
    col_fft<<<BB * WF * CC / 2, blk, 0, stream>>>(gRe, gIm, -1.f, 1.f / 256.f);
    afno_wmma_kernel<<<NTILE, blk, 0, stream>>>(gRe, gIm, w1, b1, w2, b2);
    col_fft<<<BB * WF * CC / 2, blk, 0, stream>>>(gRe, gIm, +1.f, 1.f);
    inv_row_fft<<<BB * CC * HH / 2, blk, 0, stream>>>(gRe, gIm, x, out);
}
